// PNA_85899345976
// MI455X (gfx1250) — matrix operand order repaired and checked by reading, not compiled, not run on hardware
//
#include <hip/hip_runtime.h>
#include <hip/hip_bf16.h>
#include <math.h>

// ---------------------------------------------------------------------------
// PNA forward for MI455X (gfx1250), wave32 + V_WMMA_F32_16X16X32_F16.
// All dense GEMMs run on WMMA with fp16 operands / fp32 accum.
// Row-stripe waves: one wave owns 16 rows x TN 16-col tiles, so the A
// fragment is loaded once per K-step and reused across TN WMMAs (7x less
// A-side HBM traffic; B panels are ~70 KB and stay resident in WGP$/L2).
// Tower pre-MLP is fused with the segment reduction: hs tiles stay in
// registers and are scattered with L2 atomics into 82 MB of accumulators
// (sum/sumsq/min-key/max-key), which fit in the 192 MB L2.
// Workspace requirement: ~660 MB in d_ws.
// ---------------------------------------------------------------------------

typedef __attribute__((ext_vector_type(16))) _Float16 v16h;
typedef __attribute__((ext_vector_type(8)))  _Float16 v8h;
typedef __attribute__((ext_vector_type(8)))  float    v8f;

#define TOW   5
#define FEAT  100     // F_IN == H == 100
#define FOUT  20
#define LDHID 128     // padded hidden ld (100 used)
#define LDCAT 320     // padded 3*H concat (300 used)
#define LDAGG 512     // padded T*F (500 used)
#define LDPST 1312    // padded 13*F (1300 used)
#define NPAD  112     // padded out-cols for F=100 weights

// ---- monotonic float<->uint key for atomic min/max -------------------------
__device__ __forceinline__ unsigned fkey(float f) {
  unsigned u = __float_as_uint(f);
  return (u >> 31) ? ~u : (u | 0x80000000u);
}
__device__ __forceinline__ float keyf(unsigned k) {
  return (k >> 31) ? __uint_as_float(k & 0x7fffffffu) : __uint_as_float(~k);
}

// ---------------------------------------------------------------------------
// Row-stripe WMMA GEMM: C[M,Nc] = epilogue(A[M,K] @ Bt[Nc,K]^T + bias)
// A row-major fp16 (lda), Bt = B transposed row-major fp16 (ldb), K % 32 == 0.
// One wave -> 16 rows x TN 16-col tiles; A fragment reused across TN WMMAs.
// Epilogues: bias, residual+scale, relu, fp32 and/or fp16 stores, optional
// zero-fill of tile cols >= Nc.
// ---------------------------------------------------------------------------
template <int TN>
__global__ __launch_bounds__(256)
void wmma_gemm_kernel(const _Float16* __restrict__ A, int lda,
                      const _Float16* __restrict__ Bt, int ldb,
                      const float* __restrict__ bias,
                      float* __restrict__ Cf, int ldcf,
                      _Float16* __restrict__ Ch, int ldch,
                      const float* __restrict__ resid, int ldr, float rscale,
                      int M, int Nc, int K, int relu_flag, int zero_pad)
{
  int wave = blockIdx.x * (blockDim.x >> 5) + (threadIdx.x >> 5);
  if (wave >= (M >> 4)) return;
  int lane = threadIdx.x & 31;
  int row0 = wave << 4;
  int col = lane & 15, half = lane >> 4;

  const _Float16* ap = A  + (size_t)(row0 + col) * lda + (half << 3);
  const _Float16* bp = Bt + (size_t)col * ldb + (half << 3);

  v8f acc[TN] = {};
  for (int k0 = 0; k0 < K; k0 += 32) {
    union { v16h v; v8h h[2]; } a;
    a.h[0] = *(const v8h*)(ap);
    a.h[1] = *(const v8h*)(ap + 16);
    __builtin_prefetch(ap + 64, 0, 0);   // global_prefetch_b8 two K-tiles ahead
    ap += 32;
#pragma unroll
    for (int tn = 0; tn < TN; ++tn) {
      union { v16h v; v8h h[2]; } b;
      const _Float16* bq = bp + (size_t)(tn * 16) * ldb + k0;
      b.h[0] = *(const v8h*)(bq);
      b.h[1] = *(const v8h*)(bq + 16);
      acc[tn] = __builtin_amdgcn_wmma_f32_16x16x32_f16(
          false, a.v, false, b.v, (short)0, acc[tn], false, false);
    }
  }

#pragma unroll
  for (int tn = 0; tn < TN; ++tn) {
    int c = tn * 16 + col;
#pragma unroll
    for (int r = 0; r < 8; ++r) {
      int m = row0 + r + (half << 3);
      float v;
      if (c < Nc) {
        v = acc[tn][r];
        if (bias)  v += bias[c];
        if (resid) v = resid[(size_t)m * ldr + c] + rscale * v;
        if (relu_flag) v = v > 0.f ? v : 0.f;
      } else {
        if (!zero_pad) continue;
        v = 0.f;
      }
      if (Cf) Cf[(size_t)m * ldcf + c] = v;
      if (Ch) Ch[(size_t)m * ldch + c] = (_Float16)v;
    }
  }
}

// ---------------------------------------------------------------------------
// Fused tower pre-MLP + scatter:  hs[e,t,:] row-stripe computed with WMMA
// (A fragment loaded once per K-step, 7 WMMAs) and immediately scattered
// (sum, sum^2, min-key, max-key) at node dst[e] via no-return L2 atomics.
// ---------------------------------------------------------------------------
__global__ __launch_bounds__(256)
void pre_scatter_kernel(const _Float16* __restrict__ H,      // [E, LDCAT] fp16
                        const _Float16* __restrict__ preBt,  // [TOW][NPAD][LDCAT]
                        const float* __restrict__ preB,      // [TOW, FEAT]
                        const int* __restrict__ dst,
                        float* __restrict__ s_sum, float* __restrict__ s_sq,
                        unsigned* __restrict__ s_mn, unsigned* __restrict__ s_mx,
                        int Ecnt)
{
  const int TN = (FEAT + 15) >> 4;           // 7
  int wave = blockIdx.x * (blockDim.x >> 5) + (threadIdx.x >> 5);
  if (wave >= (Ecnt >> 4) * TOW) return;
  int lane = threadIdx.x & 31;
  int t  = wave % TOW;
  int tm = wave / TOW;
  int row0 = tm << 4;
  int col = lane & 15, half = lane >> 4;

  const _Float16* ap = H + (size_t)(row0 + col) * LDCAT + (half << 3);
  const _Float16* bp = preBt + (size_t)t * NPAD * LDCAT
                             + (size_t)col * LDCAT + (half << 3);

  v8f acc[TN] = {};
  for (int k0 = 0; k0 < LDCAT; k0 += 32) {
    union { v16h v; v8h h[2]; } a;
    a.h[0] = *(const v8h*)(ap);
    a.h[1] = *(const v8h*)(ap + 16);
    __builtin_prefetch(ap + 64, 0, 0);
    ap += 32;
#pragma unroll
    for (int tn = 0; tn < TN; ++tn) {
      union { v16h v; v8h h[2]; } b;
      const _Float16* bq = bp + (size_t)(tn * 16) * LDCAT + k0;
      b.h[0] = *(const v8h*)(bq);
      b.h[1] = *(const v8h*)(bq + 16);
      acc[tn] = __builtin_amdgcn_wmma_f32_16x16x32_f16(
          false, a.v, false, b.v, (short)0, acc[tn], false, false);
    }
  }

  float biasv[TN];
#pragma unroll
  for (int tn = 0; tn < TN; ++tn) {
    int c = tn * 16 + col;
    biasv[tn] = (c < FEAT) ? preB[t * FEAT + c] : 0.f;
  }
#pragma unroll
  for (int r = 0; r < 8; ++r) {
    int e = row0 + r + (half << 3);
    int node = dst[e];                      // one index load per row
    size_t nb = (size_t)node * LDAGG + (size_t)t * FEAT;
#pragma unroll
    for (int tn = 0; tn < TN; ++tn) {
      int c = tn * 16 + col;
      if (c >= FEAT) continue;
      float val = acc[tn][r] + biasv[tn];
      size_t idx = nb + c;
      atomicAdd(&s_sum[idx], val);          // no-return L2 atomics
      atomicAdd(&s_sq[idx],  val * val);
      atomicMax(&s_mx[idx], fkey(val));
      atomicMin(&s_mn[idx], fkey(val));
    }
  }
}

// ---------------------------------------------------------------------------
// Small support kernels
// ---------------------------------------------------------------------------
__global__ void fill_f32(float* p, float v, long n) {
  long i = blockIdx.x * (long)blockDim.x + threadIdx.x;
  long s = (long)gridDim.x * blockDim.x;
  for (; i < n; i += s) p[i] = v;
}
__global__ void fill_u32(unsigned* p, unsigned v, long n) {
  long i = blockIdx.x * (long)blockDim.x + threadIdx.x;
  long s = (long)gridDim.x * blockDim.x;
  for (; i < n; i += s) p[i] = v;
}
__global__ void cvt_f16_kernel(const float* __restrict__ s, _Float16* __restrict__ d, long n) {
  long i = blockIdx.x * (long)blockDim.x + threadIdx.x;
  long st = (long)gridDim.x * blockDim.x;
  for (; i < n; i += st) d[i] = (_Float16)s[i];
}
// W[K,Nc] fp32 -> Bt[Np,Kp] fp16 transposed+padded
__global__ void prep_bt_kernel(const float* __restrict__ W, _Float16* __restrict__ Bt,
                               int K, int Nc, int Kp, int Np) {
  long i = blockIdx.x * (long)blockDim.x + threadIdx.x;
  if (i >= (long)Kp * Np) return;
  int k = (int)(i % Kp), n = (int)(i / Kp);
  float v = (n < Nc && k < K) ? W[(size_t)k * Nc + n] : 0.f;
  Bt[i] = (_Float16)v;
}
__global__ void degree_kernel(const int* __restrict__ dst, float* __restrict__ deg, int Ecnt) {
  int e = blockIdx.x * blockDim.x + threadIdx.x;
  if (e < Ecnt) atomicAdd(&deg[dst[e]], 1.f);
}
__global__ void sumlog_kernel(const float* __restrict__ deg, float* __restrict__ out, int Ncnt) {
  __shared__ float sh[256];
  float a = 0.f;
  for (long i = blockIdx.x * (long)blockDim.x + threadIdx.x; i < Ncnt;
       i += (long)gridDim.x * blockDim.x)
    a += logf(deg[i] + 1.f);
  sh[threadIdx.x] = a; __syncthreads();
  for (int o = 128; o > 0; o >>= 1) {
    if ((int)threadIdx.x < o) sh[threadIdx.x] += sh[threadIdx.x + o];
    __syncthreads();
  }
  if (threadIdx.x == 0) atomicAdd(out, sh[0]);
}
// Build edge concat buffer H[E,LDCAT].
// mode 0: [x_dst, x_src, (ea_enc left for GEMM), pad0]   (layer pre input)
// mode 1: [x_src, x_dst, ea, pad0]                       (edge-update input)
// mode 2: [relu(x_src), relu(x_dst), ea, pad0]           (readout input)
__global__ void gather_edges_kernel(const int* __restrict__ src, const int* __restrict__ dst,
                                    const float* __restrict__ x,   // [N,LDHID]
                                    const float* __restrict__ ea,  // [E,LDHID] or null
                                    _Float16* __restrict__ H, int mode, int Ecnt)
{
  long i = blockIdx.x * (long)blockDim.x + threadIdx.x;
  if (i >= (long)Ecnt * LDCAT) return;
  int e = (int)(i / LDCAT), c = (int)(i % LDCAT);
  float v = 0.f;
  if (mode == 0) {
    if (c < FEAT)            v = x[(size_t)dst[e] * LDHID + c];
    else if (c < 2 * FEAT)   v = x[(size_t)src[e] * LDHID + (c - FEAT)];
    else if (c < 3 * FEAT)   return;                 // ea_enc written by GEMM
  } else if (mode == 1) {
    if (c < FEAT)            v = x[(size_t)src[e] * LDHID + c];
    else if (c < 2 * FEAT)   v = x[(size_t)dst[e] * LDHID + (c - FEAT)];
    else if (c < 3 * FEAT)   v = ea[(size_t)e * LDHID + (c - 2 * FEAT)];
  } else {
    if (c < FEAT)            { v = x[(size_t)src[e] * LDHID + c];          v = v > 0.f ? v : 0.f; }
    else if (c < 2 * FEAT)   { v = x[(size_t)dst[e] * LDHID + (c - FEAT)]; v = v > 0.f ? v : 0.f; }
    else if (c < 3 * FEAT)   v = ea[(size_t)e * LDHID + (c - 2 * FEAT)];
  }
  H[i] = (_Float16)v;
}
// Finalize aggregators -> post-MLP input  postin[N][TOW][LDPST] fp16
__global__ void finish_aggs_kernel(const float* __restrict__ s_sum, const float* __restrict__ s_sq,
                                   const unsigned* __restrict__ s_mn, const unsigned* __restrict__ s_mx,
                                   const float* __restrict__ deg, const float* __restrict__ slog,
                                   const float* __restrict__ x,  // [N,LDHID]
                                   _Float16* __restrict__ postin, int Ncnt)
{
  long i = blockIdx.x * (long)blockDim.x + threadIdx.x;
  if (i >= (long)Ncnt * TOW * FEAT) return;
  int f = (int)(i % FEAT);
  int t = (int)((i / FEAT) % TOW);
  int n = (int)(i / ((long)TOW * FEAT));
  float d  = deg[n];
  float dc = d > 1.f ? d : 1.f;
  float avg = slog[0] / (float)Ncnt;
  float la  = logf(dc + 1.f);
  float amp = la / avg, att = avg / la;
  bool  he  = d > 0.f;
  size_t a = (size_t)n * LDAGG + (size_t)t * FEAT + f;
  float mean = s_sum[a] / dc;
  float m2   = s_sq[a]  / dc;
  float var  = m2 - mean * mean; var = var > 0.f ? var : 0.f;
  float sd   = sqrtf(var + 1e-5f);
  float mn   = he ? keyf(s_mn[a]) : 0.f;
  float mx   = he ? keyf(s_mx[a]) : 0.f;
  _Float16* row = postin + ((size_t)n * TOW + t) * LDPST;
  if (f < FEAT) row[f] = (_Float16)x[(size_t)n * LDHID + f];       // xt block
  if (f < (LDPST - 13 * FEAT)) row[13 * FEAT + f] = (_Float16)0.f; // K pad
  float av[4] = {mean, mn, mx, sd};
  float sc[3] = {1.f, amp, att};
#pragma unroll
  for (int s = 0; s < 3; ++s)
#pragma unroll
    for (int g = 0; g < 4; ++g)
      row[FEAT + (s * 4 + g) * FEAT + f] = (_Float16)(av[g] * sc[s]);
}
// training-mode batch-norm statistics per channel (biased variance)
__global__ void bn_stats_kernel(const float* __restrict__ Y, float* __restrict__ m,
                                float* __restrict__ v, int Ncnt)
{
  __shared__ float s1[256], s2[256];
  int c = blockIdx.x;
  float a = 0.f, b = 0.f;
  for (int n = threadIdx.x; n < Ncnt; n += blockDim.x) {
    float x = Y[(size_t)n * LDHID + c];
    a += x; b += x * x;
  }
  s1[threadIdx.x] = a; s2[threadIdx.x] = b; __syncthreads();
  for (int o = 128; o > 0; o >>= 1) {
    if ((int)threadIdx.x < o) { s1[threadIdx.x] += s1[threadIdx.x + o];
                                s2[threadIdx.x] += s2[threadIdx.x + o]; }
    __syncthreads();
  }
  if (threadIdx.x == 0) { float mm = s1[0] / Ncnt; m[c] = mm; v[c] = s2[0] / Ncnt - mm * mm; }
}
// x = (x + relu(bn(out)))/2, refresh fp32 hidden + fp16 copy
__global__ void bn_residual_kernel(const float* __restrict__ Y, const float* __restrict__ bm,
                                   const float* __restrict__ bv, const float* __restrict__ g,
                                   const float* __restrict__ b, float* __restrict__ x,
                                   _Float16* __restrict__ x16, int Ncnt)
{
  long i = blockIdx.x * (long)blockDim.x + threadIdx.x;
  if (i >= (long)Ncnt * LDHID) return;
  int c = (int)(i % LDHID);
  float xv = 0.f;
  if (c < FEAT) {
    float o = g[c] * (Y[i] - bm[c]) * rsqrtf(bv[c] + 1e-5f) + b[c];
    o = o > 0.f ? o : 0.f;
    xv = (x[i] + o) * 0.5f;
  }
  x[i] = xv;
  x16[i] = (_Float16)xv;
}
// final tiny projection: out[E,2] = z2[E,<=25] @ w3[25,2] + b3
__global__ void final_out_kernel(const _Float16* __restrict__ z2, const float* __restrict__ w3,
                                 const float* __restrict__ b3, float* __restrict__ out, int Ecnt)
{
  int e = blockIdx.x * blockDim.x + threadIdx.x;
  if (e >= Ecnt) return;
  float a0 = b3[0], a1 = b3[1];
#pragma unroll
  for (int k = 0; k < 25; ++k) {
    float z = (float)z2[(size_t)e * 32 + k];
    a0 += z * w3[k * 2 + 0];
    a1 += z * w3[k * 2 + 1];
  }
  out[(size_t)e * 2 + 0] = a0;
  out[(size_t)e * 2 + 1] = a1;
}

// ---------------------------------------------------------------------------
extern "C" void kernel_launch(void* const* d_in, const int* in_sizes, int n_in,
                              void* d_out, int out_size, void* d_ws, size_t ws_size,
                              hipStream_t stream)
{
  const int N = in_sizes[0] / 128;
  const int E = in_sizes[1] / 2;
  const int ED = in_sizes[2] / E;

  const float* x_in  = (const float*)d_in[0];
  const int*   ei    = (const int*)d_in[1];
  const int*   esrc  = ei;
  const int*   edst  = ei + E;
  const float* eattr = (const float*)d_in[2];
  const float* node_w = (const float*)d_in[3];
  const float* node_b = (const float*)d_in[4];
  const float* edge_w = (const float*)d_in[5];
  const float* edge_b = (const float*)d_in[6];
  const int LBASE = 7, LSTR = 14;
  const float* m_w1 = (const float*)d_in[LBASE + 2 * LSTR + 0];
  const float* m_b1 = (const float*)d_in[LBASE + 2 * LSTR + 1];
  const float* m_w2 = (const float*)d_in[LBASE + 2 * LSTR + 2];
  const float* m_b2 = (const float*)d_in[LBASE + 2 * LSTR + 3];
  const float* m_w3 = (const float*)d_in[LBASE + 2 * LSTR + 4];
  const float* m_b3 = (const float*)d_in[LBASE + 2 * LSTR + 5];

  // ---- workspace bump allocator ----
  char* base = (char*)d_ws; size_t off = 0;
  auto alloc = [&](size_t bytes) -> char* {
    off = (off + 255) & ~(size_t)255;
    char* r = base + off; off += bytes; return r;
  };
  float*    xh     = (float*)   alloc((size_t)N * LDHID * 4);
  _Float16* xh16   = (_Float16*)alloc((size_t)N * LDHID * 2);
  _Float16* x16    = (_Float16*)alloc((size_t)N * 128 * 2);
  float*    eaF    = (float*)   alloc((size_t)E * LDHID * 4);
  _Float16* ea16   = (_Float16*)alloc((size_t)E * LDHID * 2);
  _Float16* eat16  = (_Float16*)alloc((size_t)E * ED * 2);
  _Float16* H      = (_Float16*)alloc((size_t)E * LDCAT * 2);   // aliased as postin
  _Float16* t1     = (_Float16*)alloc((size_t)E * LDHID * 2);
  _Float16* z2     = (_Float16*)alloc((size_t)E * 32 * 2);
  _Float16* lin16  = (_Float16*)alloc((size_t)N * LDHID * 2);
  float*    linout = (float*)   alloc((size_t)N * LDHID * 4);
  float*    deg    = (float*)   alloc((size_t)N * 4);
  float*    slog   = (float*)   alloc(256);
  float*    bn_m   = (float*)   alloc(LDHID * 4);
  float*    bn_v   = (float*)   alloc(LDHID * 4);
  float*    acc_s  = (float*)   alloc((size_t)N * LDAGG * 4);
  float*    acc_q  = (float*)   alloc((size_t)N * LDAGG * 4);
  unsigned* acc_mn = (unsigned*)alloc((size_t)N * LDAGG * 4);
  unsigned* acc_mx = (unsigned*)alloc((size_t)N * LDAGG * 4);
  _Float16* nodeBt = (_Float16*)alloc((size_t)NPAD * 128 * 2);
  _Float16* edgeBt = (_Float16*)alloc((size_t)NPAD * 32 * 2);
  _Float16* m1Bt   = (_Float16*)alloc((size_t)64 * LDCAT * 2);
  _Float16* m2Bt   = (_Float16*)alloc((size_t)32 * 64 * 2);
  _Float16 *eeBt[2], *preBt[2], *postBt[2], *linBt[2], *em1Bt[2], *em2Bt[2];
  for (int l = 0; l < 2; ++l) {
    eeBt[l]   = (_Float16*)alloc((size_t)NPAD * 128 * 2);
    preBt[l]  = (_Float16*)alloc((size_t)TOW * NPAD * LDCAT * 2);
    postBt[l] = (_Float16*)alloc((size_t)TOW * 32 * LDPST * 2);
    linBt[l]  = (_Float16*)alloc((size_t)NPAD * 128 * 2);
    em1Bt[l]  = (_Float16*)alloc((size_t)NPAD * LDCAT * 2);
    em2Bt[l]  = (_Float16*)alloc((size_t)NPAD * 128 * 2);
  }
  _Float16* postin = H;   // alias: H is dead between pre-scatter and edge-update

  // ---- helpers ----
  auto gemm = [&](const _Float16* A, int lda, const _Float16* Bt, int ldb,
                  const float* bias, float* Cf, int ldcf, _Float16* Ch, int ldch,
                  const float* resid, int ldr, float rs,
                  int M, int Nc, int K, int relu, int zp) {
    int blocks = ((M >> 4) + 7) / 8;
    switch ((Nc + 15) >> 4) {
      case 7: wmma_gemm_kernel<7><<<blocks, 256, 0, stream>>>(
                  A, lda, Bt, ldb, bias, Cf, ldcf, Ch, ldch, resid, ldr, rs, M, Nc, K, relu, zp);
              break;
      case 4: wmma_gemm_kernel<4><<<blocks, 256, 0, stream>>>(
                  A, lda, Bt, ldb, bias, Cf, ldcf, Ch, ldch, resid, ldr, rs, M, Nc, K, relu, zp);
              break;
      case 2: wmma_gemm_kernel<2><<<blocks, 256, 0, stream>>>(
                  A, lda, Bt, ldb, bias, Cf, ldcf, Ch, ldch, resid, ldr, rs, M, Nc, K, relu, zp);
              break;
      default: wmma_gemm_kernel<1><<<blocks, 256, 0, stream>>>(
                  A, lda, Bt, ldb, bias, Cf, ldcf, Ch, ldch, resid, ldr, rs, M, Nc, K, relu, zp);
              break;
    }
  };
  auto prep = [&](const float* W, _Float16* Bt, int K, int Nc, int Kp, int Np) {
    long n = (long)Kp * Np;
    prep_bt_kernel<<<(int)((n + 255) / 256), 256, 0, stream>>>(W, Bt, K, Nc, Kp, Np);
  };

  // ---- setup: fp16 inputs, fp16 transposed weights, zero fills ----
  cvt_f16_kernel<<<2048, 256, 0, stream>>>(x_in,  x16,  (long)N * 128);
  cvt_f16_kernel<<<2048, 256, 0, stream>>>(eattr, eat16, (long)E * ED);
  prep(node_w, nodeBt, 128, FEAT, 128, NPAD);
  prep(edge_w, edgeBt, ED, FEAT, 32, NPAD);
  prep(m_w1, m1Bt, 3 * FEAT, 50, LDCAT, 64);
  prep(m_w2, m2Bt, 50, 25, 64, 32);
  for (int l = 0; l < 2; ++l) {
    const float* ee_w   = (const float*)d_in[LBASE + l * LSTR + 0];
    const float* pre_w  = (const float*)d_in[LBASE + l * LSTR + 2];
    const float* post_w = (const float*)d_in[LBASE + l * LSTR + 4];
    const float* lin_w  = (const float*)d_in[LBASE + l * LSTR + 6];
    const float* em_w1  = (const float*)d_in[LBASE + l * LSTR + 10];
    const float* em_w2  = (const float*)d_in[LBASE + l * LSTR + 12];
    prep(ee_w,  eeBt[l], FEAT, FEAT, 128, NPAD);
    prep(lin_w, linBt[l], TOW * FOUT, FEAT, 128, NPAD);
    prep(em_w1, em1Bt[l], 3 * FEAT, FEAT, LDCAT, NPAD);
    prep(em_w2, em2Bt[l], FEAT, FEAT, 128, NPAD);
    for (int t = 0; t < TOW; ++t) {
      prep(pre_w  + (size_t)t * 3 * FEAT * FEAT, preBt[l]  + (size_t)t * NPAD * LDCAT,
           3 * FEAT, FEAT, LDCAT, NPAD);
      prep(post_w + (size_t)t * 13 * FEAT * FOUT, postBt[l] + (size_t)t * 32 * LDPST,
           13 * FEAT, FOUT, LDPST, 32);
    }
  }
  fill_u32<<<2048, 256, 0, stream>>>((unsigned*)ea16,  0u, (long)E * LDHID / 2);
  fill_u32<<<2048, 256, 0, stream>>>((unsigned*)t1,    0u, (long)E * LDHID / 2);
  fill_u32<<<2048, 256, 0, stream>>>((unsigned*)lin16, 0u, (long)N * LDHID / 2);
  fill_f32<<<64, 256, 0, stream>>>(deg, 0.f, N);
  fill_f32<<<1, 64, 0, stream>>>(slog, 0.f, 1);

  // ---- degrees / avg-log ----
  degree_kernel<<<(E + 255) / 256, 256, 0, stream>>>(edst, deg, E);
  sumlog_kernel<<<64, 256, 0, stream>>>(deg, slog, N);

  // ---- encoders (WMMA) ----
  gemm(x16, 128, nodeBt, 128, node_b, xh, LDHID, xh16, LDHID,
       nullptr, 0, 0.f, N, FEAT, 128, 0, 1);
  gemm(eat16, 32, edgeBt, 32, edge_b, eaF, LDHID, ea16, LDHID,
       nullptr, 0, 0.f, E, FEAT, 32, 0, 1);

  // ---- GNN layers ----
  for (int l = 0; l < 2; ++l) {
    const float* ee_b   = (const float*)d_in[LBASE + l * LSTR + 1];
    const float* pre_b  = (const float*)d_in[LBASE + l * LSTR + 3];
    const float* post_b = (const float*)d_in[LBASE + l * LSTR + 5];
    const float* lin_b  = (const float*)d_in[LBASE + l * LSTR + 7];
    const float* bn_g   = (const float*)d_in[LBASE + l * LSTR + 8];
    const float* bn_b   = (const float*)d_in[LBASE + l * LSTR + 9];
    const float* em_b1  = (const float*)d_in[LBASE + l * LSTR + 11];
    const float* em_b2  = (const float*)d_in[LBASE + l * LSTR + 13];

    // h = [x_dst | x_src | ea_enc | 0]
    gather_edges_kernel<<<(int)(((long)E * LDCAT + 255) / 256), 256, 0, stream>>>(
        esrc, edst, xh, eaF, H, 0, E);
    gemm(ea16, LDHID, eeBt[l], 128, ee_b, nullptr, 0, H + 2 * FEAT, LDCAT,
         nullptr, 0, 0.f, E, FEAT, 128, 0, 1);

    // reset accumulators, fused pre-GEMM + scatter
    fill_f32<<<2048, 256, 0, stream>>>(acc_s, 0.f, (long)N * LDAGG);
    fill_f32<<<2048, 256, 0, stream>>>(acc_q, 0.f, (long)N * LDAGG);
    fill_u32<<<2048, 256, 0, stream>>>(acc_mn, 0xFFFFFFFFu, (long)N * LDAGG);
    fill_u32<<<2048, 256, 0, stream>>>(acc_mx, 0u,          (long)N * LDAGG);
    {
      int waves = (E >> 4) * TOW;
      pre_scatter_kernel<<<(waves + 7) / 8, 256, 0, stream>>>(
          H, preBt[l], pre_b, edst, acc_s, acc_q, acc_mn, acc_mx, E);
    }

    // finalize aggregators -> post-MLP input (aliases H)
    finish_aggs_kernel<<<(int)(((long)N * TOW * FEAT + 255) / 256), 256, 0, stream>>>(
        acc_s, acc_q, acc_mn, acc_mx, deg, slog, xh, postin, N);

    // per-tower post MLP -> lin input, then lin (WMMA)
    for (int t = 0; t < TOW; ++t)
      gemm(postin + (size_t)t * LDPST, TOW * LDPST, postBt[l] + (size_t)t * 32 * LDPST, LDPST,
           post_b + t * FOUT, nullptr, 0, lin16 + t * FOUT, LDHID,
           nullptr, 0, 0.f, N, FOUT, LDPST, 0, 0);
    gemm(lin16, LDHID, linBt[l], 128, lin_b, linout, LDHID, nullptr, 0,
         nullptr, 0, 0.f, N, FEAT, 128, 0, 0);

    // batch-norm + residual halving
    bn_stats_kernel<<<FEAT, 256, 0, stream>>>(linout, bn_m, bn_v, N);
    bn_residual_kernel<<<(int)(((long)N * LDHID + 255) / 256), 256, 0, stream>>>(
        linout, bn_m, bn_v, bn_g, bn_b, xh, xh16, N);

    // edge-update MLP: ea += 0.5 * em2(relu(em1([x_src|x_dst|ea])))
    gather_edges_kernel<<<(int)(((long)E * LDCAT + 255) / 256), 256, 0, stream>>>(
        esrc, edst, xh, eaF, H, 1, E);
    gemm(H, LDCAT, em1Bt[l], LDCAT, em_b1, nullptr, 0, t1, LDHID,
         nullptr, 0, 0.f, E, FEAT, LDCAT, 1, 1);
    gemm(t1, LDHID, em2Bt[l], 128, em_b2, eaF, LDHID, ea16, LDHID,
         eaF, LDHID, 0.5f, E, FEAT, 128, 0, 1);
  }

  // ---- readout ----
  gather_edges_kernel<<<(int)(((long)E * LDCAT + 255) / 256), 256, 0, stream>>>(
      esrc, edst, xh, eaF, H, 2, E);
  gemm(H, LDCAT, m1Bt, LDCAT, m_b1, nullptr, 0, t1, 64,
       nullptr, 0, 0.f, E, 50, LDCAT, 1, 1);
  gemm(t1, 64, m2Bt, 64, m_b2, nullptr, 0, z2, 32,
       nullptr, 0, 0.f, E, 25, 64, 1, 1);
  final_out_kernel<<<(E + 255) / 256, 256, 0, stream>>>(z2, m_w3, m_b3, (float*)d_out, E);
}
